// GNNStack_22316650070145
// MI455X (gfx1250) — matrix operand order repaired and checked by reading, not compiled, not run on hardware
//
#include <hip/hip_runtime.h>
#include <cstdint>
#include <cstddef>

typedef __attribute__((ext_vector_type(16))) _Float16 v16h;
typedef __attribute__((ext_vector_type(8)))  _Float16 v8h;
typedef __attribute__((ext_vector_type(4)))  _Float16 v4h;
typedef __attribute__((ext_vector_type(8)))  float    v8f;

#define HDIM 128   // feature width / GEMM K for every matmul in this net

// ---------------------------------------------------------------------------
// Elementwise / graph kernels
// ---------------------------------------------------------------------------

__global__ void k_fill_zero_f32(float* __restrict__ p, long long n) {
  long long i = (long long)blockIdx.x * blockDim.x + threadIdx.x;
  long long stride = (long long)gridDim.x * blockDim.x;
  for (; i < n; i += stride) p[i] = 0.0f;
}

__global__ void k_cvt_f32_f16(const float* __restrict__ in, _Float16* __restrict__ out,
                              long long n) {
  long long i = (long long)blockIdx.x * blockDim.x + threadIdx.x;
  long long stride = (long long)gridDim.x * blockDim.x;
  for (; i < n; i += stride) out[i] = (_Float16)in[i];
}

__global__ void k_count(const long long* __restrict__ dst, float* __restrict__ cnt, int E) {
  int e = blockIdx.x * blockDim.x + threadIdx.x;
  if (e < E) {
    __hip_atomic_fetch_add(&cnt[dst[e]], 1.0f, __ATOMIC_RELAXED, __HIP_MEMORY_SCOPE_AGENT);
  }
}

__global__ void k_invdeg(const float* __restrict__ cnt, float* __restrict__ inv, int N) {
  int i = blockIdx.x * blockDim.x + threadIdx.x;
  if (i < N) inv[i] = 1.0f / fmaxf(cnt[i], 1.0f);
}

// One edge per 32 threads; each lane moves 4 features (8B f16 load -> 4 f32 HW atomics).
__global__ void k_scatter_add(const _Float16* __restrict__ h,
                              const long long* __restrict__ src,
                              const long long* __restrict__ dst,
                              float* __restrict__ agg, int E) {
  long long idx = (long long)blockIdx.x * blockDim.x + threadIdx.x;
  long long e = idx >> 5;
  if (e >= E) return;
  int f = (int)(idx & 31) * 4;
  long long s = src[e];
  long long d = dst[e];
  v4h hv = *(const v4h*)(h + (size_t)s * HDIM + f);
  float* ap = agg + (size_t)d * HDIM + f;
  __hip_atomic_fetch_add(ap + 0, (float)hv[0], __ATOMIC_RELAXED, __HIP_MEMORY_SCOPE_AGENT);
  __hip_atomic_fetch_add(ap + 1, (float)hv[1], __ATOMIC_RELAXED, __HIP_MEMORY_SCOPE_AGENT);
  __hip_atomic_fetch_add(ap + 2, (float)hv[2], __ATOMIC_RELAXED, __HIP_MEMORY_SCOPE_AGENT);
  __hip_atomic_fetch_add(ap + 3, (float)hv[3], __ATOMIC_RELAXED, __HIP_MEMORY_SCOPE_AGENT);
}

// ---------------------------------------------------------------------------
// WMMA fragment loaders (wave32 layouts per CDNA5 ISA 7.12.2)
// ---------------------------------------------------------------------------

// A-matrix 16x32 f16 fragment. Row m = rowBase + lane%16.
// lanes 0-15: K = {kc..kc+7, kc+16..kc+23}; lanes 16-31: K = {kc+8.., kc+24..}
static __device__ __forceinline__ v16h load_a_f16(const _Float16* __restrict__ A,
                                                  int rowBase, int N, int kc, int lane) {
  int m = rowBase + (lane & 15);
  if (m >= N) m = N - 1;                      // per-lane clamp (cndmask, EXEC untouched)
  int base = kc + ((lane & 16) ? 8 : 0);
  const _Float16* p = A + (size_t)m * HDIM + base;
  v8h lo = *(const v8h*)(p);
  v8h hi = *(const v8h*)(p + 16);
  v16h a;
#pragma unroll
  for (int e = 0; e < 8; ++e) { a[e] = lo[e]; a[e + 8] = hi[e]; }
  return a;
}

// Same layout, but source is f32 agg scaled by per-row 1/deg (mean aggregation fused).
static __device__ __forceinline__ v16h load_a_mean(const float* __restrict__ agg,
                                                   const float* __restrict__ inv,
                                                   int rowBase, int N, int kc, int lane) {
  int m = rowBase + (lane & 15);
  if (m >= N) m = N - 1;
  float s = inv[m];
  int base = kc + ((lane & 16) ? 8 : 0);
  const float* p = agg + (size_t)m * HDIM + base;
  v8f lo = *(const v8f*)(p);
  v8f hi = *(const v8f*)(p + 16);
  v16h a;
#pragma unroll
  for (int e = 0; e < 8; ++e) {
    a[e]     = (_Float16)(lo[e] * s);
    a[e + 8] = (_Float16)(hi[e] * s);
  }
  return a;
}

// B-matrix 32x16 f16 fragment from row-major W[out, in]:
// col n = colBase + lane%16, K = {kc+8h..kc+8h+7, kc+16+8h..kc+16+8h+7} (same as A).
static __device__ __forceinline__ v16h load_b(const _Float16* __restrict__ W,
                                              int colBase, int kc, int lane) {
  int n = colBase + (lane & 15);
  int k0 = kc + ((lane & 16) ? 8 : 0);
  const _Float16* p = W + (size_t)n * HDIM + k0;
  v8h lo = *(const v8h*)(p);
  v8h hi = *(const v8h*)(p + 16);
  v16h b;
#pragma unroll
  for (int e = 0; e < 8; ++e) { b[e] = lo[e]; b[e + 8] = hi[e]; }
  return b;
}

static __device__ __forceinline__ v8f wmma_f16(v16h a, v16h b, v8f c) {
  return __builtin_amdgcn_wmma_f32_16x16x32_f16(false, a, false, b, (short)0, c,
                                                false, false);
}

// ---------------------------------------------------------------------------
// Fused SAGEConv GEMM:  hout = relu( mean @ Wl^T + bl + hin @ Wr^T )
// block = 256 (8 waves, one 16-col tile each), grid.x = ceil(N/16)
// ---------------------------------------------------------------------------
__global__ void k_sage_gemm(const float* __restrict__ agg, const float* __restrict__ inv,
                            const _Float16* __restrict__ hin,
                            const _Float16* __restrict__ Wl, const float* __restrict__ bl,
                            const _Float16* __restrict__ Wr,
                            _Float16* __restrict__ hout, int N) {
  int rowBase = blockIdx.x * 16;
  int lane = threadIdx.x & 31;
  int colBase = (threadIdx.x >> 5) * 16;   // 8 waves cover Hout=128

  v8f acc = {};
#pragma unroll
  for (int kc = 0; kc < HDIM; kc += 32) {
    v16h am = load_a_mean(agg, inv, rowBase, N, kc, lane);
    v16h ah = load_a_f16(hin, rowBase, N, kc, lane);
    v16h bL = load_b(Wl, colBase, kc, lane);
    v16h bR = load_b(Wr, colBase, kc, lane);
    acc = wmma_f16(am, bL, acc);
    acc = wmma_f16(ah, bR, acc);
  }

  int n = colBase + (lane & 15);
  float bias = bl[n];
  int mBase = rowBase + ((lane & 16) ? 8 : 0);
#pragma unroll
  for (int r = 0; r < 8; ++r) {
    int m = mBase + r;
    if (m < N) {
      float v = acc[r] + bias;
      v = fmaxf(v, 0.0f);
      hout[(size_t)m * HDIM + n] = (_Float16)v;
    }
  }
}

// ---------------------------------------------------------------------------
// Plain GEMM + bias:  out[N, Hout] = A[N,128] @ W[Hout,128]^T + b
// block = 32 * (Hout/16), grid.x = ceil(N/16)
// ---------------------------------------------------------------------------
template <typename OutT>
__global__ void k_gemm_bias(const _Float16* __restrict__ A, const _Float16* __restrict__ W,
                            const float* __restrict__ bias, OutT* __restrict__ out,
                            int N, int Hout) {
  int rowBase = blockIdx.x * 16;
  int lane = threadIdx.x & 31;
  int colBase = (threadIdx.x >> 5) * 16;

  v8f acc = {};
#pragma unroll
  for (int kc = 0; kc < HDIM; kc += 32) {
    v16h a = load_a_f16(A, rowBase, N, kc, lane);
    v16h b = load_b(W, colBase, kc, lane);
    acc = wmma_f16(a, b, acc);
  }

  int n = colBase + (lane & 15);
  float bv = bias[n];
  int mBase = rowBase + ((lane & 16) ? 8 : 0);
#pragma unroll
  for (int r = 0; r < 8; ++r) {
    int m = mBase + r;
    if (m < N) out[(size_t)m * Hout + n] = (OutT)(acc[r] + bv);
  }
}

// ---------------------------------------------------------------------------
// In-place log_softmax over rows of 64 (one wave32 per row, 2 cols per lane)
// ---------------------------------------------------------------------------
__global__ void k_logsoftmax64(float* __restrict__ out, int N) {
  int row = blockIdx.x * (blockDim.x >> 5) + (threadIdx.x >> 5);
  if (row >= N) return;
  int lane = threadIdx.x & 31;
  float* p = out + (size_t)row * 64;
  float a = p[lane];
  float b = p[lane + 32];
  float mx = fmaxf(a, b);
#pragma unroll
  for (int m = 16; m >= 1; m >>= 1) mx = fmaxf(mx, __shfl_xor(mx, m, 32));
  float s = expf(a - mx) + expf(b - mx);
#pragma unroll
  for (int m = 16; m >= 1; m >>= 1) s += __shfl_xor(s, m, 32);
  float lse = mx + logf(s);
  p[lane]      = a - lse;
  p[lane + 32] = b - lse;
}

// ---------------------------------------------------------------------------
// Launch
// ---------------------------------------------------------------------------
extern "C" void kernel_launch(void* const* d_in, const int* in_sizes, int n_in,
                              void* d_out, int out_size, void* d_ws, size_t ws_size,
                              hipStream_t stream) {
  const float*     x   = (const float*)d_in[0];
  const long long* ei  = (const long long*)d_in[1];
  const float* Wl[3] = { (const float*)d_in[2], (const float*)d_in[5], (const float*)d_in[8] };
  const float* bl[3] = { (const float*)d_in[3], (const float*)d_in[6], (const float*)d_in[9] };
  const float* Wr[3] = { (const float*)d_in[4], (const float*)d_in[7], (const float*)d_in[10] };
  const float* Wp1 = (const float*)d_in[11];
  const float* bp1 = (const float*)d_in[12];
  const float* Wp2 = (const float*)d_in[13];
  const float* bp2 = (const float*)d_in[14];

  const int N = in_sizes[0] / HDIM;        // 50000 (multiple of 16)
  const int E = in_sizes[1] / 2;           // 800000
  const int O = in_sizes[13] / HDIM;       // 64

  const long long* srcIdx = ei;
  const long long* dstIdx = ei + E;

  // ---- workspace carve-out (256B aligned) ----
  char* wsp = (char*)d_ws;
  auto carve = [&](size_t bytes) -> char* {
    char* p = wsp;
    wsp += (bytes + 255) & ~(size_t)255;
    return p;
  };
  _Float16* hA   = (_Float16*)carve((size_t)N * HDIM * sizeof(_Float16));
  _Float16* hB   = (_Float16*)carve((size_t)N * HDIM * sizeof(_Float16));
  float*    agg  = (float*)carve((size_t)N * HDIM * sizeof(float));
  float*    cnt  = (float*)carve((size_t)N * sizeof(float));
  float*    inv  = (float*)carve((size_t)N * sizeof(float));
  _Float16* w16  = (_Float16*)carve((size_t)(7 * HDIM * HDIM + O * HDIM) * sizeof(_Float16));
  _Float16* Wlh[3], *Wrh[3];
  {
    _Float16* q = w16;
    for (int l = 0; l < 3; ++l) { Wlh[l] = q; q += HDIM * HDIM; Wrh[l] = q; q += HDIM * HDIM; }
  }
  _Float16* Wp1h = w16 + 6 * HDIM * HDIM;
  _Float16* Wp2h = Wp1h + HDIM * HDIM;
  _Float16* p16  = (_Float16*)agg;         // head intermediate aliases agg (free by then)

  const int TB = 256;
  auto blocks = [](long long n, int tb) { return (unsigned)((n + tb - 1) / tb); };

  // ---- one-time-per-call prep: f16 weights, f16 input features, degrees ----
  k_cvt_f32_f16<<<blocks((long long)N * HDIM, TB), TB, 0, stream>>>(x, hA, (long long)N * HDIM);
  for (int l = 0; l < 3; ++l) {
    k_cvt_f32_f16<<<blocks(HDIM * HDIM, TB), TB, 0, stream>>>(Wl[l], Wlh[l], HDIM * HDIM);
    k_cvt_f32_f16<<<blocks(HDIM * HDIM, TB), TB, 0, stream>>>(Wr[l], Wrh[l], HDIM * HDIM);
  }
  k_cvt_f32_f16<<<blocks(HDIM * HDIM, TB), TB, 0, stream>>>(Wp1, Wp1h, HDIM * HDIM);
  k_cvt_f32_f16<<<blocks((long long)O * HDIM, TB), TB, 0, stream>>>(Wp2, Wp2h, (long long)O * HDIM);

  k_fill_zero_f32<<<blocks(N, TB), TB, 0, stream>>>(cnt, N);
  k_count<<<blocks(E, TB), TB, 0, stream>>>(dstIdx, cnt, E);
  k_invdeg<<<blocks(N, TB), TB, 0, stream>>>(cnt, inv, N);

  // ---- 3 SAGE layers (ping-pong hA <-> hB) ----
  _Float16* hin = hA;
  _Float16* hout = hB;
  const unsigned gemmGrid = blocks(N, 16);
  for (int l = 0; l < 3; ++l) {
    k_fill_zero_f32<<<blocks((long long)N * HDIM, TB), TB, 0, stream>>>(agg, (long long)N * HDIM);
    k_scatter_add<<<blocks((long long)E * 32, TB), TB, 0, stream>>>(hin, srcIdx, dstIdx, agg, E);
    k_sage_gemm<<<gemmGrid, 256, 0, stream>>>(agg, inv, hin, Wlh[l], bl[l], Wrh[l], hout, N);
    _Float16* t = hin; hin = hout; hout = t;
  }

  // ---- head: Linear(128,128) -> Linear(128,64) -> log_softmax ----
  k_gemm_bias<_Float16><<<gemmGrid, 256, 0, stream>>>(hin, Wp1h, bp1, p16, N, HDIM);
  k_gemm_bias<float><<<gemmGrid, 32 * (O / 16), 0, stream>>>(p16, Wp2h, bp2, (float*)d_out, N, O);
  k_logsoftmax64<<<blocks(N, 8), 256, 0, stream>>>((float*)d_out, N);
}
